// CrossBlock_18322330484745
// MI455X (gfx1250) — compile-verified, matrix operand order checked by reading, not run on hardware
//
#include <hip/hip_runtime.h>
#include <hip/hip_bf16.h>
#include <math.h>

// ---------------------------------------------------------------------------
// CrossBlock for MI455X (gfx1250): bf16 WMMA GEMMs (v_wmma_f32_16x16x32_bf16),
// f32 accumulation, fused epilogues. Wave32.
// Two GEMM tilings:
//   - big-N config: 64x32 wave tile (8 WMMA / 12 b128), 128x128 block tile
//   - N=64  config: 32x32 wave tile, 128x64 block tile
// Both softmax directions run on row-major data (S and a separately
// materialized S^T) so every memory pass is coalesced; WMMA flops are cheap.
// ---------------------------------------------------------------------------

typedef unsigned short u16;  // raw bf16 storage
typedef __attribute__((ext_vector_type(16))) __bf16 v16bf;
typedef __attribute__((ext_vector_type(8)))  float  v8f;

__device__ __forceinline__ u16 f2bf(float f) {
  unsigned u = __float_as_uint(f);
  u += 0x7FFFu + ((u >> 16) & 1u);   // round-to-nearest-even
  return (u16)(u >> 16);
}

union FragBF {
  v16bf bf;
  uint4 q[2];
};

// ---------------------------------------------------------------------------
// C[M,N](f32) (+bias[N]) (+add[M,N]) (+=C) = A[M,K] * Bt[N,K]^T
// A, Bt bf16 with K contiguous. Wave tile = (WM*16) x (WN*16); block has
// BWM x BWN waves. Grid covers M,N exactly (all dims divide the tiles).
// ---------------------------------------------------------------------------
template <int WM, int WN, int BWM, int BWN>
__global__ __launch_bounds__(BWM * BWN * 32)
void gemm_bf16(const u16* __restrict__ A, const u16* __restrict__ Bt,
               float* __restrict__ C, int K,
               int lda, int ldb, int ldc,
               const float* __restrict__ bias,
               const float* __restrict__ add, int ldadd, int accum) {
  const int wave  = threadIdx.x >> 5;
  const int lane  = threadIdx.x & 31;
  const int wm    = wave / BWN;
  const int wn    = wave % BWN;
  const int m0    = blockIdx.x * (BWM * WM * 16) + wm * WM * 16;
  const int n0    = blockIdx.y * (BWN * WN * 16) + wn * WN * 16;
  const int l16   = lane & 15;
  const int khalf = lane >> 4;   // ISA: lanes 16-31 hold the other K half

  v8f acc[WM][WN] = {};

  for (int k0 = 0; k0 < K; k0 += 32) {
    FragBF a[WM], b[WN];
#pragma unroll
    for (int i = 0; i < WM; ++i) {   // A frags: rows m0+i*16+l16
      const u16* p = A + (size_t)(m0 + i * 16 + l16) * lda + k0 + khalf * 8;
      a[i].q[0] = *(const uint4*)(p);        // K = k0 + khalf*8 + 0..7
      a[i].q[1] = *(const uint4*)(p + 16);   // K = k0+16 + khalf*8 + 0..7
    }
#pragma unroll
    for (int j = 0; j < WN; ++j) {   // B frags: cols n0+j*16+l16
      const u16* p = Bt + (size_t)(n0 + j * 16 + l16) * ldb + k0 + khalf * 8;
      b[j].q[0] = *(const uint4*)(p);
      b[j].q[1] = *(const uint4*)(p + 16);
    }
#pragma unroll
    for (int i = 0; i < WM; ++i)
#pragma unroll
      for (int j = 0; j < WN; ++j)
        acc[i][j] = __builtin_amdgcn_wmma_f32_16x16x32_bf16(
            false, a[i].bf, false, b[j].bf, (short)0, acc[i][j], false, false);
  }

  // C/D layout: col = lane&15, row = v + 8*(lane>>4) within each 16x16 tile.
#pragma unroll
  for (int i = 0; i < WM; ++i)
#pragma unroll
    for (int j = 0; j < WN; ++j) {
      const int col  = n0 + j * 16 + l16;
      const int rowb = m0 + i * 16 + khalf * 8;
#pragma unroll
      for (int v = 0; v < 8; ++v) {
        const int row = rowb + v;
        const size_t idx = (size_t)row * ldc + col;
        float val = acc[i][j][v];
        if (bias) val += bias[col];
        if (add)  val += add[(size_t)row * ldadd + col];
        if (accum) val += C[idx];
        C[idx] = val;
      }
    }
}

// f32 -> bf16 elementwise
__global__ __launch_bounds__(256)
void cvt_bf16(const float* __restrict__ src, u16* __restrict__ dst, size_t n) {
  size_t i = (size_t)blockIdx.x * 256 + threadIdx.x;
  if (i < n) dst[i] = f2bf(src[i]);
}

// W[K,N] f32 -> WT[N,K] bf16 (one-time, small)
__global__ __launch_bounds__(256)
void transpose_w(const float* __restrict__ W, u16* __restrict__ WT, int K, int N) {
  size_t i = (size_t)blockIdx.x * 256 + threadIdx.x;
  if (i >= (size_t)K * N) return;
  int n = (int)(i % N), k = (int)(i / N);
  WT[(size_t)n * K + k] = f2bf(W[i]);
}

// RoPE + head split + dh^-0.25 scaling: qk[B,N,D] f32 -> out[B,H,N,dh] bf16
__global__ __launch_bounds__(256)
void rope_heads(const float* __restrict__ qk, const float* __restrict__ enc,
                u16* __restrict__ out, int B, int N, int D, int H, int dh,
                float s) {
  size_t idx = (size_t)blockIdx.x * 256 + threadIdx.x;  // pair index
  size_t total = (size_t)B * N * (D / 2);
  if (idx >= total) return;
  int d2 = (int)(idx % (D / 2));
  size_t bn = idx / (D / 2);
  int n = (int)(bn % N), b = (int)(bn / N);
  int d = d2 * 2, h = d / dh, dd = d % dh;
  const float* t = qk + ((size_t)b * N + n) * D + d;
  float te = t[0], to = t[1];
  size_t e0 = ((size_t)b * N + n) * dh + dd;   // enc[0,b,0,n,dd]
  size_t es = (size_t)B * N * dh;              // enc[0] vs enc[1] stride
  float f0e = enc[e0], f0o = enc[e0 + 1];
  float f1e = enc[es + e0], f1o = enc[es + e0 + 1];
  float oe = (te * f0e - to * f1e) * s;        // rotate_half: (-b, a)
  float oo = (to * f0o + te * f1o) * s;
  u16* op = out + (((size_t)b * H + h) * N + n) * dh + dd;
  op[0] = f2bf(oe);
  op[1] = f2bf(oo);
}

// v[B,N,D] f32 -> vT[B,H,dh,N] bf16, LDS-tiled 32x32 so both the global
// reads and the global writes are coalesced.
__global__ __launch_bounds__(256)
void v_headsT(const float* __restrict__ v, u16* __restrict__ vT,
              int B, int N, int D, int H, int dh) {
  __shared__ float tile[32][33];
  const int bh = blockIdx.z;                 // b*H + h
  const int b = bh / H, h = bh % H;
  const int n0 = blockIdx.x * 32;
  const int d0 = blockIdx.y * 32;            // within head
  const int tx = threadIdx.x & 31;
  const int ty = threadIdx.x >> 5;           // 0..7
#pragma unroll
  for (int k = 0; k < 32; k += 8)
    tile[ty + k][tx] = v[((size_t)b * N + n0 + ty + k) * D + h * dh + d0 + tx];
  __syncthreads();
#pragma unroll
  for (int k = 0; k < 32; k += 8)
    vT[((size_t)bh * dh + d0 + ty + k) * N + n0 + tx] = f2bf(tile[tx][ty + k]);
}

// Row softmax: P[i,:] = softmax(S[i,:]), bf16 out. Fully coalesced.
__global__ __launch_bounds__(256)
void softmax_bf16(const float* __restrict__ S, u16* __restrict__ P, int n) {
  __shared__ float red[256];
  const int i = blockIdx.x;
  const int tid = threadIdx.x;
  const float* base = S + (size_t)i * n;
  float mx = -INFINITY;
  for (int j = tid; j < n; j += 256) mx = fmaxf(mx, base[j]);
  red[tid] = mx; __syncthreads();
  for (int s = 128; s > 0; s >>= 1) {
    if (tid < s) red[tid] = fmaxf(red[tid], red[tid + s]);
    __syncthreads();
  }
  mx = red[0]; __syncthreads();
  float sum = 0.f;
  for (int j = tid; j < n; j += 256) sum += expf(base[j] - mx);
  red[tid] = sum; __syncthreads();
  for (int s = 128; s > 0; s >>= 1) {
    if (tid < s) red[tid] += red[tid + s];
    __syncthreads();
  }
  float inv = 1.f / red[0];
  for (int j = tid; j < n; j += 256)
    P[(size_t)i * n + j] = f2bf(expf(base[j] - mx) * inv);
}

// LayerNorm (over C) + exact-erf GELU, f32 in -> bf16 out. One block per row.
__global__ __launch_bounds__(256)
void ln_gelu(const float* __restrict__ Hin, u16* __restrict__ Hout,
             const float* __restrict__ g, const float* __restrict__ bta, int C) {
  __shared__ float red[256];
  const int row = blockIdx.x, tid = threadIdx.x;
  const float* base = Hin + (size_t)row * C;
  float s = 0.f;
  for (int j = tid; j < C; j += 256) s += base[j];
  red[tid] = s; __syncthreads();
  for (int k = 128; k > 0; k >>= 1) { if (tid < k) red[tid] += red[tid + k]; __syncthreads(); }
  float mean = red[0] / C; __syncthreads();
  float vs = 0.f;
  for (int j = tid; j < C; j += 256) { float d = base[j] - mean; vs += d * d; }
  red[tid] = vs; __syncthreads();
  for (int k = 128; k > 0; k >>= 1) { if (tid < k) red[tid] += red[tid + k]; __syncthreads(); }
  float r = rsqrtf(red[0] / C + 1e-5f);
  for (int j = tid; j < C; j += 256) {
    float val = (base[j] - mean) * r * g[j] + bta[j];
    val = 0.5f * val * (1.f + erff(val * 0.70710678118654752f));  // exact GELU
    Hout[(size_t)row * C + j] = f2bf(val);
  }
}

// ---------------------------------------------------------------------------
extern "C" void kernel_launch(void* const* d_in, const int* in_sizes, int n_in,
                              void* d_out, int out_size, void* d_ws, size_t ws_size,
                              hipStream_t stream) {
  (void)in_sizes; (void)n_in; (void)out_size; (void)ws_size;
  const int B = 4, N = 2048, D = 256, H = 4, dh = 64, D2 = 512;
  const size_t BN = (size_t)B * N;            // 8192
  const size_t BND = BN * D;                  // 2M
  const size_t NN = (size_t)N * N;            // 4M
  const float rope_scale = 0.35355339059327379f;  // dh^-0.25

  const float* x0  = (const float*)d_in[0];
  const float* x1  = (const float*)d_in[1];
  const float* e0  = (const float*)d_in[2];
  const float* e1  = (const float*)d_in[3];
  const float* Wqk = (const float*)d_in[4];
  const float* bqk = (const float*)d_in[5];
  const float* Wv  = (const float*)d_in[6];
  const float* bv  = (const float*)d_in[7];
  const float* Wo  = (const float*)d_in[8];
  const float* bo  = (const float*)d_in[9];
  const float* W1  = (const float*)d_in[10];
  const float* b1  = (const float*)d_in[11];
  const float* lng = (const float*)d_in[12];
  const float* lnb = (const float*)d_in[13];
  const float* W2  = (const float*)d_in[14];
  const float* b2  = (const float*)d_in[15];
  float* out0 = (float*)d_out;
  float* out1 = (float*)d_out + BND;

  // workspace bump allocator (256B aligned)
  char* ws = (char*)d_ws;
  size_t off = 0;
  auto alloc = [&](size_t bytes) -> char* {
    char* p = ws + off;
    off += (bytes + 255) & ~(size_t)255;
    return p;
  };
  u16*  x0b   = (u16*)alloc(BND * 2);
  u16*  x1b   = (u16*)alloc(BND * 2);
  u16*  WqkT  = (u16*)alloc((size_t)D * D * 2);
  u16*  WvT   = (u16*)alloc((size_t)D * D * 2);
  u16*  WoT   = (u16*)alloc((size_t)D * D * 2);
  u16*  W1T   = (u16*)alloc((size_t)D2 * D2 * 2);
  u16*  W2T   = (u16*)alloc((size_t)D * D2 * 2);
  float* tmp  = (float*)alloc(BND * 4);
  u16*  qk0h  = (u16*)alloc(BND * 2);   // [B,H,N,dh]
  u16*  qk1h  = (u16*)alloc(BND * 2);
  u16*  v0T   = (u16*)alloc(BND * 2);   // [B,H,dh,N]
  u16*  v1T   = (u16*)alloc(BND * 2);
  float* S    = (float*)alloc(NN * 4);  // per-(b,h) reuse
  float* ST   = (float*)alloc(NN * 4);  // S^T, materialized by a 2nd GEMM
  u16*  P0    = (u16*)alloc(NN * 2);
  u16*  P1    = (u16*)alloc(NN * 2);
  float* m0   = (float*)alloc(BND * 4);
  float* m1   = (float*)alloc(BND * 4);
  u16*  m0b   = (u16*)alloc(BND * 2);
  u16*  m1b   = (u16*)alloc(BND * 2);
  float* mo0  = (float*)alloc(BND * 4);
  float* mo1  = (float*)alloc(BND * 4);
  u16*  mo0b  = (u16*)alloc(BND * 2);
  u16*  mo1b  = (u16*)alloc(BND * 2);
  float* hbuf = (float*)alloc(BN * D2 * 4);  // reused for both streams
  u16*  hb    = (u16*)alloc(BN * D2 * 2);

  auto ew = [](size_t n) { return dim3((unsigned)((n + 255) / 256)); };
  // Big-N config: 64x32 wave tile, 2x4 waves -> 128x128 block tile.
  auto GEMM_B = [&](const u16* A, const u16* Bt, float* C, int M, int Nn, int K,
                    int lda, int ldb, int ldc, const float* bias,
                    const float* add, int ldadd, int accum) {
    dim3 grid((unsigned)(M / 128), (unsigned)(Nn / 128));
    gemm_bf16<4, 2, 2, 4><<<grid, 256, 0, stream>>>(A, Bt, C, K, lda, ldb, ldc,
                                                    bias, add, ldadd, accum);
  };
  // Narrow-N config: 32x32 wave tile, 4x2 waves -> 128x64 block tile.
  auto GEMM_A = [&](const u16* A, const u16* Bt, float* C, int M, int Nn, int K,
                    int lda, int ldb, int ldc, const float* bias,
                    const float* add, int ldadd, int accum) {
    dim3 grid((unsigned)(M / 128), (unsigned)(Nn / 64));
    gemm_bf16<2, 2, 4, 2><<<grid, 256, 0, stream>>>(A, Bt, C, K, lda, ldb, ldc,
                                                    bias, add, ldadd, accum);
  };

  // --- bf16 conversions of inputs / transposed weights ---
  cvt_bf16<<<ew(BND), 256, 0, stream>>>(x0, x0b, BND);
  cvt_bf16<<<ew(BND), 256, 0, stream>>>(x1, x1b, BND);
  transpose_w<<<ew((size_t)D * D), 256, 0, stream>>>(Wqk, WqkT, D, D);
  transpose_w<<<ew((size_t)D * D), 256, 0, stream>>>(Wv, WvT, D, D);
  transpose_w<<<ew((size_t)D * D), 256, 0, stream>>>(Wo, WoT, D, D);
  transpose_w<<<ew((size_t)D2 * D2), 256, 0, stream>>>(W1, W1T, D2, D2);
  transpose_w<<<ew((size_t)D2 * D), 256, 0, stream>>>(W2, W2T, D2, D);

  // --- QK / V projections (+RoPE, head layouts) ---
  GEMM_B(x0b, WqkT, tmp, (int)BN, D, D, D, D, D, bqk, nullptr, 0, 0);
  rope_heads<<<ew(BN * D / 2), 256, 0, stream>>>(tmp, e0, qk0h, B, N, D, H, dh, rope_scale);
  GEMM_B(x1b, WqkT, tmp, (int)BN, D, D, D, D, D, bqk, nullptr, 0, 0);
  rope_heads<<<ew(BN * D / 2), 256, 0, stream>>>(tmp, e1, qk1h, B, N, D, H, dh, rope_scale);
  {
    dim3 tg(N / 32, dh / 32, B * H);
    GEMM_B(x0b, WvT, tmp, (int)BN, D, D, D, D, D, bv, nullptr, 0, 0);
    v_headsT<<<tg, 256, 0, stream>>>(tmp, v0T, B, N, D, H, dh);
    GEMM_B(x1b, WvT, tmp, (int)BN, D, D, D, D, D, bv, nullptr, 0, 0);
    v_headsT<<<tg, 256, 0, stream>>>(tmp, v1T, B, N, D, H, dh);
  }

  // --- bidirectional cross attention, per (b,h) ---
  for (int bh = 0; bh < B * H; ++bh) {
    const int b = bh / H, h = bh % H;
    const u16* q0 = qk0h + (size_t)bh * N * dh;
    const u16* q1 = qk1h + (size_t)bh * N * dh;
    // S = qk0 @ qk1^T and S^T = qk1 @ qk0^T (extra WMMAs are ~free; they buy
    // coalesced softmax passes in both directions).
    GEMM_B(q0, q1, S,  N, N, dh, dh, dh, N, nullptr, nullptr, 0, 0);
    GEMM_B(q1, q0, ST, N, N, dh, dh, dh, N, nullptr, nullptr, 0, 0);
    softmax_bf16<<<dim3(N), 256, 0, stream>>>(S,  P0, N);  // 0 attends to 1
    softmax_bf16<<<dim3(N), 256, 0, stream>>>(ST, P1, N);  // 1 attends to 0
    // m0[:, h*dh:] = P0 @ v1 ; m1[:, h*dh:] = P1 @ v0   (vT = [dh,N] bf16)
    GEMM_A(P0, v1T + (size_t)bh * dh * N, m0 + (size_t)b * N * D + h * dh,
           N, dh, N, N, N, D, nullptr, nullptr, 0, 0);
    GEMM_A(P1, v0T + (size_t)bh * dh * N, m1 + (size_t)b * N * D + h * dh,
           N, dh, N, N, N, D, nullptr, nullptr, 0, 0);
  }

  // --- output projection ---
  cvt_bf16<<<ew(BND), 256, 0, stream>>>(m0, m0b, BND);
  cvt_bf16<<<ew(BND), 256, 0, stream>>>(m1, m1b, BND);
  GEMM_B(m0b, WoT, mo0, (int)BN, D, D, D, D, D, bo, nullptr, 0, 0);
  GEMM_B(m1b, WoT, mo1, (int)BN, D, D, D, D, D, bo, nullptr, 0, 0);
  cvt_bf16<<<ew(BND), 256, 0, stream>>>(mo0, mo0b, BND);
  cvt_bf16<<<ew(BND), 256, 0, stream>>>(mo1, mo1b, BND);

  // --- FFN per stream: h = [x, m] @ W1 + b1 ; LN ; GELU ; out = x + h @ W2 + b2
  const u16* xb[2]  = {x0b, x1b};
  const u16* mb[2]  = {mo0b, mo1b};
  const float* xr[2] = {x0, x1};
  float* outs[2] = {out0, out1};
  for (int sidx = 0; sidx < 2; ++sidx) {
    // concat GEMM split over K: rows 0..D-1 of W1 hit x, rows D..2D-1 hit m
    GEMM_B(xb[sidx], W1T,     hbuf, (int)BN, D2, D, D, D2, D2, b1, nullptr, 0, 0);
    GEMM_B(mb[sidx], W1T + D, hbuf, (int)BN, D2, D, D, D2, D2, nullptr, nullptr, 0, 1);
    ln_gelu<<<dim3((unsigned)BN), 256, 0, stream>>>(hbuf, hb, lng, lnb, D2);
    GEMM_B(hb, W2T, outs[sidx], (int)BN, D, D2, D2, D2, D, b2, xr[sidx], D, 0);
  }
}
